// AdaptiveRankLinear_49460843381303
// MI455X (gfx1250) — matrix operand order repaired and checked by reading, not compiled, not run on hardware
//
#include <hip/hip_runtime.h>

typedef _Float16 v16h __attribute__((ext_vector_type(16)));
typedef _Float16 v8h  __attribute__((ext_vector_type(8)));
typedef float    v8f  __attribute__((ext_vector_type(8)));

#define IN_F    1024
#define OUT_F   1024
#define BATCH_N 16384
#define RNK     20

// ---------------------------------------------------------------------------
// Kernel 1: exact f32 TT factor chains.
//   L[k][r]  (1024 x 20):  left chain over input cores  (c0..c3)
//   R[r][o]  (20 x 1024):  right chain over output cores (c4..c7)
// Each thread owns one k (or one o) and carries a 20-float rank vector.
// ---------------------------------------------------------------------------
__global__ __launch_bounds__(256) void tt_factors(
    const float* __restrict__ c0, const float* __restrict__ c1,
    const float* __restrict__ c2, const float* __restrict__ c3,
    const float* __restrict__ c4, const float* __restrict__ c5,
    const float* __restrict__ c6, const float* __restrict__ c7,
    float* __restrict__ L, float* __restrict__ R)
{
    int tid = blockIdx.x * 256 + threadIdx.x;
    if (tid < IN_F) {
        int k  = tid;
        int i3 = k & 3, i2 = (k >> 2) & 7, i1 = (k >> 5) & 7, i0 = (k >> 8) & 3;
        float w1[RNK], w2[RNK], w3[RNK];
        #pragma unroll
        for (int r = 0; r < RNK; ++r) w1[r] = c0[i0 * RNK + r];          // (1,4,20)
        #pragma unroll
        for (int r2 = 0; r2 < RNK; ++r2) {
            float s = 0.f;
            #pragma unroll
            for (int r1 = 0; r1 < RNK; ++r1)
                s += w1[r1] * c1[(r1 * 8 + i1) * RNK + r2];              // (20,8,20)
            w2[r2] = s;
        }
        #pragma unroll
        for (int r3 = 0; r3 < RNK; ++r3) {
            float s = 0.f;
            #pragma unroll
            for (int r2 = 0; r2 < RNK; ++r2)
                s += w2[r2] * c2[(r2 * 8 + i2) * RNK + r3];              // (20,8,20)
            w3[r3] = s;
        }
        #pragma unroll
        for (int r4 = 0; r4 < RNK; ++r4) {
            float s = 0.f;
            #pragma unroll
            for (int r3 = 0; r3 < RNK; ++r3)
                s += w3[r3] * c3[(r3 * 4 + i3) * RNK + r4];              // (20,4,20)
            L[k * RNK + r4] = s;
        }
    } else if (tid < 2 * OUT_F) {
        int o  = tid - OUT_F;
        int o3 = o & 3, o2 = (o >> 2) & 7, o1 = (o >> 5) & 7, o0 = (o >> 8) & 3;
        float v7[RNK], v6[RNK], v5[RNK];
        #pragma unroll
        for (int r7 = 0; r7 < RNK; ++r7) v7[r7] = c7[r7 * 4 + o3];      // (20,4,1)
        #pragma unroll
        for (int r6 = 0; r6 < RNK; ++r6) {
            float s = 0.f;
            #pragma unroll
            for (int r7 = 0; r7 < RNK; ++r7)
                s += c6[(r6 * 8 + o2) * RNK + r7] * v7[r7];              // (20,8,20)
            v6[r6] = s;
        }
        #pragma unroll
        for (int r5 = 0; r5 < RNK; ++r5) {
            float s = 0.f;
            #pragma unroll
            for (int r6 = 0; r6 < RNK; ++r6)
                s += c5[(r5 * 8 + o1) * RNK + r6] * v6[r6];              // (20,8,20)
            v5[r5] = s;
        }
        #pragma unroll
        for (int p = 0; p < RNK; ++p) {
            float s = 0.f;
            #pragma unroll
            for (int r5 = 0; r5 < RNK; ++r5)
                s += c4[(p * 4 + o0) * RNK + r5] * v5[r5];               // (20,4,20)
            R[p * OUT_F + o] = s;
        }
    }
}

// ---------------------------------------------------------------------------
// Kernel 2: Wt[o][k] = f16( sum_r L[k][r] * R[r][o] )   (transposed weight)
// Per block, o is constant (R values broadcast); k is the fast index.
// ---------------------------------------------------------------------------
__global__ __launch_bounds__(256) void tt_materialize(
    const float* __restrict__ L, const float* __restrict__ R,
    _Float16* __restrict__ Wt)
{
    int e = blockIdx.x * 256 + threadIdx.x;      // 0 .. 1024*1024-1
    int k = e & (IN_F - 1);
    int o = e >> 10;
    float acc = 0.f;
    #pragma unroll
    for (int r = 0; r < RNK; ++r)
        acc += L[k * RNK + r] * R[r * OUT_F + o];
    Wt[(size_t)o * IN_F + k] = (_Float16)acc;
}

// ---------------------------------------------------------------------------
// Kernel 3: out[b][o] = sum_k f16(x[b][k]) * Wt[o][k] + bias[o]
// 256 threads = 8 waves. Macro-tile 128(M) x 64(N), BK = 32.
// Wave grid 4x2, each wave owns a 32x32 region = 2x2 WMMA 16x16 tiles.
// LDS rows padded to 40 halfs (80 B) -> 16B-aligned b128 fragment loads.
// Fragment layouts follow CDNA5 ISA §7.12.2 (wave32).
// ---------------------------------------------------------------------------
#define BM  128
#define BN  64
#define BK  32
#define LDA 40

__global__ __launch_bounds__(256) void tt_gemm(
    const float* __restrict__ x, const _Float16* __restrict__ Wt,
    const float* __restrict__ bias, float* __restrict__ out)
{
    __shared__ __align__(16) _Float16 As[BM * LDA];  // x tile  (f16)
    __shared__ __align__(16) _Float16 Bs[BN * LDA];  // Wt tile (f16)

    const int tid   = threadIdx.x;
    const int lane  = tid & 31;
    const int wave  = tid >> 5;        // 0..7
    const int waveM = wave >> 1;       // 0..3  (32-row strip)
    const int waveN = wave & 1;        // 0..1  (32-col strip)

    const int bBase = blockIdx.y * BM;
    const int oBase = blockIdx.x * BN;

    // cooperative staging indices
    const int xr = tid >> 1;           // 0..127 : x row in tile
    const int xk = (tid & 1) * 16;     // 0 / 16 : k offset (16 floats)
    const int wc = tid >> 2;           // 0..63  : Wt row (output col)
    const int wk = (tid & 3) * 8;      // 0/8/16/24 : k offset (8 halfs)

    // fragment lane decomposition (A: m, B/D: n)
    const int lo15 = lane & 15;
    const int g    = lane >> 4;        // lane-group 0/1

    v8f acc[2][2] = {};                // f32 accumulators

    for (int k0 = 0; k0 < IN_F; k0 += BK) {
        // ---- stage x (f32 global -> f16 LDS) : 16 floats/thread ----
        const float4* xp =
            reinterpret_cast<const float4*>(x + (size_t)(bBase + xr) * IN_F + k0 + xk);
        float4 f0 = xp[0], f1 = xp[1], f2 = xp[2], f3 = xp[3];
        v8h h0, h1;
        h0[0] = (_Float16)f0.x; h0[1] = (_Float16)f0.y;
        h0[2] = (_Float16)f0.z; h0[3] = (_Float16)f0.w;
        h0[4] = (_Float16)f1.x; h0[5] = (_Float16)f1.y;
        h0[6] = (_Float16)f1.z; h0[7] = (_Float16)f1.w;
        h1[0] = (_Float16)f2.x; h1[1] = (_Float16)f2.y;
        h1[2] = (_Float16)f2.z; h1[3] = (_Float16)f2.w;
        h1[4] = (_Float16)f3.x; h1[5] = (_Float16)f3.y;
        h1[6] = (_Float16)f3.z; h1[7] = (_Float16)f3.w;
        *reinterpret_cast<v8h*>(&As[xr * LDA + xk])     = h0;
        *reinterpret_cast<v8h*>(&As[xr * LDA + xk + 8]) = h1;

        // ---- stage Wt (f16 global -> LDS) : 8 halfs/thread ----
        v8h wv = *reinterpret_cast<const v8h*>(
            Wt + (size_t)(oBase + wc) * IN_F + k0 + wk);
        *reinterpret_cast<v8h*>(&Bs[wc * LDA + wk]) = wv;

        __syncthreads();

        // ---- fragments ----
        // A 16x32 f16: lane(g,lo15) -> row m=lo15, K = g*8+0..7 then 16+g*8+0..7
        // B 32x16 f16: lane(g,lo15) -> col n=lo15, K = g*8+0..7 then 16+g*8+0..7
        union F { v16h v; v8h h[2]; };
        F a0, a1, b0, b1;
        {
            int ab0 = (waveM * 32 + lo15) * LDA + g * 8;
            int ab1 = ab0 + 16 * LDA;
            a0.h[0] = *reinterpret_cast<const v8h*>(&As[ab0]);
            a0.h[1] = *reinterpret_cast<const v8h*>(&As[ab0 + 16]);
            a1.h[0] = *reinterpret_cast<const v8h*>(&As[ab1]);
            a1.h[1] = *reinterpret_cast<const v8h*>(&As[ab1 + 16]);
            int bb0 = (waveN * 32 + lo15) * LDA + g * 8;
            int bb1 = bb0 + 16 * LDA;
            b0.h[0] = *reinterpret_cast<const v8h*>(&Bs[bb0]);
            b0.h[1] = *reinterpret_cast<const v8h*>(&Bs[bb0 + 16]);
            b1.h[0] = *reinterpret_cast<const v8h*>(&Bs[bb1]);
            b1.h[1] = *reinterpret_cast<const v8h*>(&Bs[bb1 + 16]);
        }

        acc[0][0] = __builtin_amdgcn_wmma_f32_16x16x32_f16(
            false, a0.v, false, b0.v, (short)0, acc[0][0], false, false);
        acc[0][1] = __builtin_amdgcn_wmma_f32_16x16x32_f16(
            false, a0.v, false, b1.v, (short)0, acc[0][1], false, false);
        acc[1][0] = __builtin_amdgcn_wmma_f32_16x16x32_f16(
            false, a1.v, false, b0.v, (short)0, acc[1][0], false, false);
        acc[1][1] = __builtin_amdgcn_wmma_f32_16x16x32_f16(
            false, a1.v, false, b1.v, (short)0, acc[1][1], false, false);

        __syncthreads();
    }

    // ---- epilogue: C/D layout: VGPR j -> M = g*8 + j, N = lo15 ----
    #pragma unroll
    for (int mt = 0; mt < 2; ++mt) {
        int rowBase = bBase + waveM * 32 + mt * 16 + g * 8;
        #pragma unroll
        for (int nt = 0; nt < 2; ++nt) {
            int col = oBase + waveN * 32 + nt * 16 + lo15;
            float bv = bias[col];
            #pragma unroll
            for (int j = 0; j < 8; ++j) {
                out[(size_t)(rowBase + j) * OUT_F + col] = acc[mt][nt][j] + bv;
            }
        }
    }
}

// ---------------------------------------------------------------------------
extern "C" void kernel_launch(void* const* d_in, const int* in_sizes, int n_in,
                              void* d_out, int out_size, void* d_ws, size_t ws_size,
                              hipStream_t stream) {
    const float* x    = (const float*)d_in[0];
    const float* c0   = (const float*)d_in[1];
    const float* c1   = (const float*)d_in[2];
    const float* c2   = (const float*)d_in[3];
    const float* c3   = (const float*)d_in[4];
    const float* c4   = (const float*)d_in[5];
    const float* c5   = (const float*)d_in[6];
    const float* c6   = (const float*)d_in[7];
    const float* c7   = (const float*)d_in[8];
    const float* bias = (const float*)d_in[9];
    float* out = (float*)d_out;

    // workspace layout: Wt f16 (2 MB) | L f32 (80 KB) | R f32 (80 KB)
    char* ws = (char*)d_ws;
    _Float16* Wt = (_Float16*)ws;
    float* L = (float*)(ws + (size_t)IN_F * OUT_F * sizeof(_Float16));
    float* R = (float*)(ws + (size_t)IN_F * OUT_F * sizeof(_Float16)
                           + (size_t)IN_F * RNK * sizeof(float));

    tt_factors<<<8, 256, 0, stream>>>(c0, c1, c2, c3, c4, c5, c6, c7, L, R);
    tt_materialize<<<(IN_F * OUT_F) / 256, 256, 0, stream>>>(L, R, Wt);

    dim3 grid(OUT_F / BN, BATCH_N / BM);   // (16, 128)
    tt_gemm<<<grid, 256, 0, stream>>>(x, Wt, bias, out);
}
